// MultimodalMultiheadAttention_83348135346807
// MI455X (gfx1250) — matrix operand order repaired and checked by reading, not compiled, not run on hardware
//
#include <hip/hip_runtime.h>

#define E_   256
#define H_   8
#define HD_  32
#define T_   768
#define B_   4
#define MR_  (T_ * B_)   // 3072 rows

typedef __attribute__((ext_vector_type(16))) _Float16 v16h;
typedef __attribute__((ext_vector_type(8)))  float    v8f;
typedef int v4i_t __attribute__((vector_size(16)));

union Frag16 { v16h v; uint4 q[2]; };

// ---- CDNA5 async global->LDS path (guarded; sync fallback keeps compile safe)
#if defined(__has_builtin)
#  if __has_builtin(__builtin_amdgcn_global_load_async_to_lds_b128)
#    define HAVE_ASYNC_LDS 1
#  endif
#endif

#if defined(HAVE_ASYNC_LDS)
#  if __has_builtin(__builtin_amdgcn_s_wait_asynccnt)
#    define WAIT_ASYNC0() __builtin_amdgcn_s_wait_asynccnt(0)
#  else
#    define WAIT_ASYNC0() asm volatile("s_wait_asynccnt 0x0" ::: "memory")
#  endif
#else
#  define WAIT_ASYNC0()
#endif

// copy 32 contiguous bytes global -> LDS (two b128 transfers per lane)
__device__ __forceinline__ void copy32B_to_lds(const _Float16* g, _Float16* l) {
#if defined(HAVE_ASYNC_LDS)
  __builtin_amdgcn_global_load_async_to_lds_b128((v4i_t*)(g + 0), (v4i_t*)(l + 0), 0, 0);
  __builtin_amdgcn_global_load_async_to_lds_b128((v4i_t*)(g + 8), (v4i_t*)(l + 8), 0, 0);
#else
  uint4 a = *(const uint4*)g;
  uint4 b = *(const uint4*)(g + 8);
  *(uint4*)l = a;
  *(uint4*)(l + 8) = b;
#endif
}

__device__ __forceinline__ v8f vzero8() {
  v8f z = {0.f, 0.f, 0.f, 0.f, 0.f, 0.f, 0.f, 0.f};
  return z;
}

__device__ __forceinline__ v8f wmma16(v16h a, v16h b, v8f c) {
  // D = A(16x32 f16) * B(32x16 f16) + C(16x16 f32)
  return __builtin_amdgcn_wmma_f32_16x16x32_f16(false, a, false, b, (short)0, c,
                                                false, false);
}

// A-fragment: 16 rows x 32 K from row-major, K-contiguous matrix.
// lane holds row (lane&15); K run of 16 contiguous halves at hi*16.
__device__ __forceinline__ v16h load_fragA(const _Float16* p, int ldk) {
  int lane = threadIdx.x & 31;
  const _Float16* rp = p + (size_t)(lane & 15) * ldk + ((lane >> 4) << 4);
  Frag16 f;
  f.q[0] = *(const uint4*)rp;
  f.q[1] = *(const uint4*)(rp + 8);
  return f.v;
}

// B-fragment: K=32 x N=16, from (N x K) row-major matrix (i.e. computes A*Brm^T).
// lane holds column (lane&15); K run of 16 contiguous halves at hi*16.
__device__ __forceinline__ v16h load_fragB(const _Float16* p, int ldk) {
  int lane = threadIdx.x & 31;
  const _Float16* rp = p + (size_t)(lane & 15) * ldk + ((lane >> 4) << 4);
  Frag16 f;
  f.q[0] = *(const uint4*)rp;
  f.q[1] = *(const uint4*)(rp + 8);
  return f.v;
}

// ---------------------------------------------------------------------------
// Generic WMMA GEMM (NT): C[m,n] = alpha*(sum_k A[m,k]*B[n,k] + bias[n]) [+C]
// block = 128 threads (4 waves); tile 64x64; grid = (N/64, M/64)
// B tile (64x32 f16, shared by all 4 waves) is staged through LDS with
// double-buffered async global->LDS copies overlapping the WMMAs.
// ---------------------------------------------------------------------------
__global__ void k_gemm_nt(const _Float16* __restrict__ A, int lda,
                          const _Float16* __restrict__ Bm, int ldb,
                          float* __restrict__ Cf, _Float16* __restrict__ Ch,
                          int ldc, int K, float alpha,
                          const float* __restrict__ bias, int accumulate) {
  __shared__ __align__(16) _Float16 Bs[2][64 * 32];
  int tid = threadIdx.x;
  int lane = tid & 31, wave = tid >> 5;
  int r = lane & 15, hi = lane >> 4;
  int rowBase = blockIdx.y * 64 + wave * 16;
  int colBase = blockIdx.x * 64;

  // cooperative B-tile copy mapping: each thread owns 32B of a 4KB tile
  int cpRow = tid >> 1;            // 0..63
  int cpOff = (tid & 1) << 4;      // 0 or 16 halves
  const _Float16* Bg = Bm + (size_t)(colBase + cpRow) * ldb + cpOff;

  v8f acc0 = vzero8(), acc1 = vzero8(), acc2 = vzero8(), acc3 = vzero8();
  const _Float16* Ap = A + (size_t)rowBase * lda;

  // prologue: stage k=0 B tile into buffer 0
  copy32B_to_lds(Bg, &Bs[0][cpRow * 32 + cpOff]);
  WAIT_ASYNC0();
  __syncthreads();

  v16h av = load_fragA(Ap, lda);
  int p = 0;
  for (int k0 = 0; k0 < K; k0 += 32) {
    bool more = (k0 + 32) < K;
    v16h avn = av;
    if (more) {
      // async-stage next B tile into the other buffer (overlaps WMMAs below)
      copy32B_to_lds(Bg + k0 + 32, &Bs[p ^ 1][cpRow * 32 + cpOff]);
      avn = load_fragA(Ap + k0 + 32, lda);
      if (k0 + 64 < K) __builtin_prefetch(Ap + (size_t)r * lda + k0 + 64, 0, 1);
    }

    // consume current buffer from LDS (ds_load_b128 fragments)
    const _Float16* bp = &Bs[p][0];
    v16h b0 = load_fragB(bp + 0 * 16 * 32, 32);
    v16h b1 = load_fragB(bp + 1 * 16 * 32, 32);
    v16h b2 = load_fragB(bp + 2 * 16 * 32, 32);
    v16h b3 = load_fragB(bp + 3 * 16 * 32, 32);
    acc0 = wmma16(av, b0, acc0);
    acc1 = wmma16(av, b1, acc1);
    acc2 = wmma16(av, b2, acc2);
    acc3 = wmma16(av, b3, acc3);

    if (more) {
      WAIT_ASYNC0();
      __syncthreads();
      av = avn;
      p ^= 1;
    }
  }

  int m0 = rowBase + hi * 8;
#pragma unroll
  for (int nt = 0; nt < 4; nt++) {
    v8f a = (nt == 0) ? acc0 : (nt == 1) ? acc1 : (nt == 2) ? acc2 : acc3;
    int n = colBase + nt * 16 + r;
    float bv = bias ? bias[n] : 0.f;
#pragma unroll
    for (int rr = 0; rr < 8; rr++) {
      size_t idx = (size_t)(m0 + rr) * ldc + n;
      float v = alpha * (a[rr] + bv);
      if (Cf) Cf[idx] = accumulate ? (Cf[idx] + v) : v;
      else    Ch[idx] = (_Float16)v;
    }
  }
}

// ---------------------------------------------------------------------------
// Segmented flash attention. One pass over columns per segment {L,A,V},
// maintaining (m_c, s_c, o_c) online; all 7 branch outputs reconstructed.
// block = 128 (4 waves, 16 q-rows each); grid = (T/64, B*H)
// ---------------------------------------------------------------------------
__global__ void k_flash(const _Float16* __restrict__ Qh,
                        const _Float16* __restrict__ Kh,
                        const _Float16* __restrict__ Vt,
                        _Float16* __restrict__ Abuf) {
  __shared__ __align__(16) _Float16 Ptile[4][16 * 32];
  int lane = threadIdx.x & 31, wave = threadIdx.x >> 5;
  int r = lane & 15, hi = lane >> 4;
  int bh = blockIdx.y, b = bh >> 3, h = bh & 7;
  int t0 = blockIdx.x * 64 + wave * 16;
  const int ldq = B_ * E_;  // 1024 halves between consecutive t/s rows

  v16h qf = load_fragA(Qh + (size_t)(t0 * B_ + b) * E_ + h * HD_, ldq);

  float mseg[3][8], sseg[3][8];
  v8f oacc[3][2];
#pragma unroll
  for (int c = 0; c < 3; c++) {
#pragma unroll
    for (int rr = 0; rr < 8; rr++) { mseg[c][rr] = -1e30f; sseg[c][rr] = 0.f; }
    oacc[c][0] = vzero8();
    oacc[c][1] = vzero8();
  }

  const int segLo[3] = {0, 296, 544};     // 8-aligned loop starts (16B loads)
  const int segBg[3] = {0, 300, 550};     // true segment bounds
  const int segEn[3] = {300, 550, 768};
  const _Float16* kb  = Kh + (size_t)b * E_ + h * HD_;
  const _Float16* vtb = Vt + (size_t)bh * HD_ * T_;
  _Float16* Pw = &Ptile[wave][0];

#pragma unroll
  for (int c = 0; c < 3; c++) {
    int sB = segBg[c], sE = segEn[c];
    for (int s0 = segLo[c]; s0 < sE; s0 += 32) {
      // ---- scores: two 16x16 tiles, K = HD = 32 in one WMMA each ----
      v16h kf0, kf1;
      {
        const _Float16* rp = kb + (size_t)(s0 + r) * ldq + (hi << 4);
        Frag16 f; f.q[0] = *(const uint4*)rp; f.q[1] = *(const uint4*)(rp + 8);
        kf0 = f.v;
        rp += (size_t)16 * ldq;
        Frag16 g; g.q[0] = *(const uint4*)rp; g.q[1] = *(const uint4*)(rp + 8);
        kf1 = g.v;
      }
      v8f sc0 = wmma16(qf, kf0, vzero8());
      v8f sc1 = wmma16(qf, kf1, vzero8());

      int col0 = s0 + r, col1 = s0 + 16 + r;
      bool ok0 = (col0 >= sB) && (col0 < sE);
      bool ok1 = (col1 >= sB) && (col1 < sE);

      // ---- online softmax update per row (rows split across halves) ----
#pragma unroll
      for (int rr = 0; rr < 8; rr++) {
        float x0 = ok0 ? sc0[rr] : -1e30f;
        float x1 = ok1 ? sc1[rr] : -1e30f;
        float mx = fmaxf(x0, x1);
#pragma unroll
        for (int d = 1; d < 16; d <<= 1) mx = fmaxf(mx, __shfl_xor(mx, d, 32));
        float mold = mseg[c][rr];
        float mnew = fmaxf(mold, mx);
        float scl  = __expf(mold - mnew);
        float e0 = ok0 ? __expf(x0 - mnew) : 0.f;
        float e1 = ok1 ? __expf(x1 - mnew) : 0.f;
        float rs = e0 + e1;
#pragma unroll
        for (int d = 1; d < 16; d <<= 1) rs += __shfl_xor(rs, d, 32);
        sseg[c][rr] = sseg[c][rr] * scl + rs;
        mseg[c][rr] = mnew;
        oacc[c][0][rr] *= scl;
        oacc[c][1][rr] *= scl;
        int prow = rr + (hi << 3);
        Pw[prow * 32 + r]      = (_Float16)e0;   // C-layout -> LDS 16x32 tile
        Pw[prow * 32 + 16 + r] = (_Float16)e1;
      }
      asm volatile("s_wait_dscnt 0" ::: "memory");

      // ---- O += P @ V (A-frag from LDS, B-frags from hd-major V) ----
      v16h pf = load_fragA(Pw, 32);
#pragma unroll
      for (int nt = 0; nt < 2; nt++) {
        const _Float16* vp = vtb + (size_t)(nt * 16 + r) * T_ + s0 + (hi << 4);
        Frag16 f; f.q[0] = *(const uint4*)vp; f.q[1] = *(const uint4*)(vp + 8);
        oacc[c][nt] = wmma16(pf, f.v, oacc[c][nt]);
      }
      asm volatile("s_wait_dscnt 0" ::: "memory");
    }
  }

  // ---- reconstruct the 7 branch outputs per row and store f16 ----
  const int segsub[7] = {7, 3, 5, 6, 1, 2, 4};  // lav,la,lv,av,l,a,v
#pragma unroll
  for (int rr = 0; rr < 8; rr++) {
    int t = t0 + rr + (hi << 3);
    int st = (t < 300) ? 0 : ((t < 550) ? 1 : 2);
#pragma unroll
    for (int i = 0; i < 7; i++) {
      float a0 = 0.f, a1 = 0.f;
      if ((segsub[i] >> st) & 1) {
        float Mx = -1e30f;
#pragma unroll
        for (int c = 0; c < 3; c++)
          if ((segsub[i] >> c) & 1) Mx = fmaxf(Mx, mseg[c][rr]);
        float den = 0.f;
#pragma unroll
        for (int c = 0; c < 3; c++)
          if ((segsub[i] >> c) & 1) {
            float w = __expf(mseg[c][rr] - Mx);
            den += sseg[c][rr] * w;
            a0 += w * oacc[c][0][rr];
            a1 += w * oacc[c][1][rr];
          }
        float inv = 1.f / den;
        a0 *= inv;
        a1 *= inv;
      }
      _Float16* dst = Abuf + (size_t)i * MR_ * E_ +
                      (size_t)(t * B_ + b) * E_ + h * HD_;
      dst[r]      = (_Float16)a0;
      dst[16 + r] = (_Float16)a1;
    }
  }
}

// ---------------------------------------------------------------------------
// Small helper kernels
// ---------------------------------------------------------------------------
__global__ void k_f32_to_f16(const float* __restrict__ s, _Float16* __restrict__ d, int n) {
  int i = blockIdx.x * 256 + threadIdx.x;
  if (i < n) d[i] = (_Float16)s[i];
}

__global__ void k_transp_f32_f16(const float* __restrict__ s, _Float16* __restrict__ d) {
  int i = blockIdx.x * 256 + threadIdx.x;          // 256x256
  int rr = i >> 8, cc = i & 255;
  d[(size_t)rr * 256 + cc] = (_Float16)s[(size_t)cc * 256 + rr];
}

__global__ void k_transp_f16(const _Float16* __restrict__ s, _Float16* __restrict__ d) {
  int i = blockIdx.x * 256 + threadIdx.x;
  int rr = i >> 8, cc = i & 255;
  d[(size_t)rr * 256 + cc] = s[(size_t)cc * 256 + rr];
}

// GT_i[n,k] = sum of up to three s_*_w[n, off+k] slices (rows stride 4E)
__global__ void k_sum_slices(_Float16* __restrict__ dst,
                             const float* s0, int o0,
                             const float* s1, int o1,
                             const float* s2, int o2) {
  int i = blockIdx.x * 256 + threadIdx.x;
  int nrow = i >> 8, k = i & 255;
  float v = s0[(size_t)nrow * 1024 + o0 + k];
  if (s1) v += s1[(size_t)nrow * 1024 + o1 + k];
  if (s2) v += s2[(size_t)nrow * 1024 + o2 + k];
  dst[i] = (_Float16)v;
}

// Vt[bh][hd][s] <- Vh[(s*B+b)*E + h*HD + hd]
__global__ void k_vt(const _Float16* __restrict__ Vh, _Float16* __restrict__ Vt) {
  int i = blockIdx.x * 256 + threadIdx.x;
  if (i >= 32 * HD_ * T_) return;
  int s = i % T_;
  int hh = i / T_;
  int hd = hh & 31, bh = hh >> 5;
  int b = bh >> 3, h = bh & 7;
  Vt[i] = Vh[(size_t)(s * B_ + b) * E_ + h * HD_ + hd];
}

// c0[n] = sum_i out_b[i] @ G_i  + (s_l_b + s_a_b + s_v_b)   (G_i[k,n] = GT_i[n,k])
__global__ void k_c0(const float* __restrict__ out_b, const _Float16* __restrict__ GT,
                     const float* __restrict__ lb, const float* __restrict__ ab,
                     const float* __restrict__ vb, float* __restrict__ c0) {
  int n = threadIdx.x;
  float acc = lb[n] + ab[n] + vb[n];
  for (int i = 0; i < 7; i++) {
    const _Float16* g = GT + (size_t)i * 65536 + (size_t)n * 256;
    const float* ob = out_b + i * 256;
    for (int k = 0; k < 256; k++) acc += ob[k] * (float)g[k];
  }
  c0[n] = acc;
}

// const_out[n] = c0 @ final_w.T + final_b
__global__ void k_constout(const float* __restrict__ c0, const float* __restrict__ fw,
                           const float* __restrict__ fb, float* __restrict__ co) {
  int n = threadIdx.x;
  float acc = fb[n];
  const float* row = fw + (size_t)n * 256;
  for (int k = 0; k < 256; k++) acc += c0[k] * row[k];
  co[n] = acc;
}

// ---------------------------------------------------------------------------
extern "C" void kernel_launch(void* const* d_in, const int* in_sizes, int n_in,
                              void* d_out, int out_size, void* d_ws, size_t ws_size,
                              hipStream_t stream) {
  (void)in_sizes; (void)n_in; (void)out_size; (void)ws_size;
  const float* q_in      = (const float*)d_in[0];
  const float* k_in      = (const float*)d_in[1];
  const float* v_in      = (const float*)d_in[2];
  const float* in_proj_w = (const float*)d_in[3];
  const float* in_proj_b = (const float*)d_in[4];
  const float* out_w     = (const float*)d_in[5];
  const float* out_b     = (const float*)d_in[6];
  const float* s_l_w     = (const float*)d_in[7];
  const float* s_l_b     = (const float*)d_in[8];
  const float* s_a_w     = (const float*)d_in[9];
  const float* s_a_b     = (const float*)d_in[10];
  const float* s_v_w     = (const float*)d_in[11];
  const float* s_v_b     = (const float*)d_in[12];
  const float* final_w   = (const float*)d_in[13];
  const float* final_b   = (const float*)d_in[14];
  float* out = (float*)d_out;

  char* p = (char*)d_ws;
  auto alloc = [&](size_t bytes) {
    char* r = p;
    p += (bytes + 255) & ~(size_t)255;
    return r;
  };
  const size_t actB = (size_t)MR_ * E_ * sizeof(_Float16);  // 1.5 MB
  const size_t matB = (size_t)E_ * E_ * sizeof(_Float16);   // 128 KB
  _Float16* Xq   = (_Float16*)alloc(actB);
  _Float16* Xk   = (_Float16*)alloc(actB);
  _Float16* Xv   = (_Float16*)alloc(actB);
  _Float16* W16  = (_Float16*)alloc((size_t)3 * E_ * E_ * 2);
  _Float16* Qhh  = (_Float16*)alloc(actB);
  _Float16* Khh  = (_Float16*)alloc(actB);
  _Float16* Vhh  = (_Float16*)alloc(actB);
  _Float16* Vtb  = (_Float16*)alloc((size_t)32 * HD_ * T_ * 2);
  _Float16* Abuf = (_Float16*)alloc(7 * actB);
  _Float16* GT   = (_Float16*)alloc(7 * matB);
  _Float16* Gt2  = (_Float16*)alloc(7 * matB);
  _Float16* owT  = (_Float16*)alloc(7 * matB);
  _Float16* Rb   = (_Float16*)alloc(7 * matB);
  _Float16* Qm   = (_Float16*)alloc(7 * matB);
  _Float16* FW16 = (_Float16*)alloc(matB);
  float* c0vec   = (float*)alloc(E_ * sizeof(float));
  float* coVec   = (float*)alloc(E_ * sizeof(float));

  const float scaling = 0.17677669529663687f;  // 1/sqrt(HD)

  // 1) convert activations + weights to f16
  int n = MR_ * E_;
  k_f32_to_f16<<<(n + 255) / 256, 256, 0, stream>>>(q_in, Xq, n);
  k_f32_to_f16<<<(n + 255) / 256, 256, 0, stream>>>(k_in, Xk, n);
  k_f32_to_f16<<<(n + 255) / 256, 256, 0, stream>>>(v_in, Xv, n);
  n = 3 * E_ * E_;
  k_f32_to_f16<<<(n + 255) / 256, 256, 0, stream>>>(in_proj_w, W16, n);
  n = E_ * E_;
  k_f32_to_f16<<<(n + 255) / 256, 256, 0, stream>>>(final_w, FW16, n);

  // 2) QKV projections (q folded with 1/sqrt(HD))
  dim3 gAct(E_ / 64, MR_ / 64);  // (4, 48)
  k_gemm_nt<<<gAct, 128, 0, stream>>>(Xq, E_, W16, E_, nullptr, Qhh, E_, E_,
                                      scaling, in_proj_b, 0);
  k_gemm_nt<<<gAct, 128, 0, stream>>>(Xk, E_, W16 + (size_t)E_ * E_, E_, nullptr,
                                      Khh, E_, E_, 1.f, in_proj_b + E_, 0);
  k_gemm_nt<<<gAct, 128, 0, stream>>>(Xv, E_, W16 + (size_t)2 * E_ * E_, E_, nullptr,
                                      Vhh, E_, E_, 1.f, in_proj_b + 2 * E_, 0);

  // 3) repack V to [bh][hd][s] for contiguous B-fragments in P@V
  n = 32 * HD_ * T_;
  k_vt<<<(n + 255) / 256, 256, 0, stream>>>(Vhh, Vtb);

  // 4) build fused weight matrices: G_i (as GT), then Q_i = final_w @ G_i^T @ out_w[i]
  auto sum3 = [&](int i, const float* a, int oa, const float* b2, int ob2,
                  const float* c, int oc) {
    k_sum_slices<<<256, 256, 0, stream>>>(GT + (size_t)i * E_ * E_, a, oa, b2, ob2, c, oc);
  };
  sum3(0, s_l_w, 0,   s_a_w, 0,   s_v_w, 0);
  sum3(1, s_l_w, 256, s_a_w, 256, nullptr, 0);
  sum3(2, s_l_w, 512, s_v_w, 256, nullptr, 0);
  sum3(3, s_a_w, 512, s_v_w, 512, nullptr, 0);
  sum3(4, s_l_w, 768, nullptr, 0, nullptr, 0);
  sum3(5, s_a_w, 768, nullptr, 0, nullptr, 0);
  sum3(6, s_v_w, 768, nullptr, 0, nullptr, 0);

  dim3 gMat(E_ / 64, E_ / 64);  // (4, 4)
  for (int i = 0; i < 7; i++) {
    size_t mo = (size_t)i * E_ * E_;
    k_transp_f16<<<256, 256, 0, stream>>>(GT + mo, Gt2 + mo);
    k_transp_f32_f16<<<256, 256, 0, stream>>>(out_w + mo, owT + mo);
    // R_i = final_w @ G_i^T
    k_gemm_nt<<<gMat, 128, 0, stream>>>(FW16, E_, Gt2 + mo, E_, nullptr, Rb + mo,
                                        E_, E_, 1.f, nullptr, 0);
    // Q_i = R_i @ out_w[i]  (= P_i^T)
    k_gemm_nt<<<gMat, 128, 0, stream>>>(Rb + mo, E_, owT + mo, E_, nullptr, Qm + mo,
                                        E_, E_, 1.f, nullptr, 0);
  }

  // 5) constant bias: const_out = (sum_i out_b[i]@G_i + s_*_b sums) @ final_w.T + final_b
  k_c0<<<1, 256, 0, stream>>>(out_b, GT, s_l_b, s_a_b, s_v_b, c0vec);
  k_constout<<<1, 256, 0, stream>>>(c0vec, final_w, final_b, coVec);

  // 6) segmented flash attention -> 7 branch activation buffers
  k_flash<<<dim3(T_ / 64, B_ * H_), 128, 0, stream>>>(Qhh, Khh, Vtb, Abuf);

  // 7) y = sum_i a_i @ P_i + const_out  (accumulated into d_out, f32)
  for (int i = 0; i < 7; i++) {
    k_gemm_nt<<<gAct, 128, 0, stream>>>(Abuf + (size_t)i * MR_ * E_, E_,
                                        Qm + (size_t)i * E_ * E_, E_,
                                        out, nullptr, E_, E_, 1.f,
                                        (i == 0) ? coVec : nullptr,
                                        (i == 0) ? 0 : 1);
  }
}
